// MultiScaleRetention_49203145343565
// MI455X (gfx1250) — matrix operand order repaired and checked by reading, not compiled, not run on hardware
//
#include <hip/hip_runtime.h>
#include <cstdint>

// ---------------------------------------------------------------------------
// MultiScaleRetention for MI455X (gfx1250): bf16 WMMA everywhere, fp32 epilogues
// B=1, S=2048, D=2048, H=16, HD=128
// ---------------------------------------------------------------------------

typedef __attribute__((ext_vector_type(16))) __bf16 v16bf;
typedef __attribute__((ext_vector_type(8)))  __bf16 v8bf;
typedef __attribute__((ext_vector_type(8)))  float  v8f;
typedef __attribute__((ext_vector_type(4)))  int    v4i;

#define S_LEN 2048
#define DIM   2048
#define NHEAD 16
#define HDIM  128

// ---- gfx1250 async global->LDS copies (ASYNCcnt), guarded ------------------
#if defined(__HIP_DEVICE_COMPILE__) &&                              \
    __has_builtin(__builtin_amdgcn_global_load_async_to_lds_b128) && \
    __has_builtin(__builtin_amdgcn_s_wait_asynccnt)
#define HAS_ASYNC_LDS 1
#else
#define HAS_ASYNC_LDS 0
#endif

// 16-byte global -> LDS copy (async when available).
// Builtin params are generic v4i* on this toolchain (per diagnostic); the
// compiler inserts the needed addrspacecasts.
__device__ __forceinline__ void lds_cp16(void* lds_dst, const void* g_src) {
#if HAS_ASYNC_LDS
  __builtin_amdgcn_global_load_async_to_lds_b128((v4i*)g_src, (v4i*)lds_dst, 0, 0);
#else
  *(v8bf*)lds_dst = *(const v8bf*)g_src;
#endif
}

__device__ __forceinline__ void async_wait() {
#if HAS_ASYNC_LDS
  __builtin_amdgcn_s_wait_asynccnt(0);
#endif
}

// build a 16-elem bf16 fragment from two 16B LDS/global chunks
__device__ __forceinline__ v16bf frag16(const __bf16* p0, const __bf16* p1) {
  v8bf lo = *(const v8bf*)p0;
  v8bf hi = *(const v8bf*)p1;
  v16bf r;
#pragma unroll
  for (int i = 0; i < 8; ++i) { r[i] = lo[i]; r[i + 8] = hi[i]; }
  return r;
}

// ---------------------------------------------------------------------------
// fp32 -> bf16 cast (4 elems/thread)
// ---------------------------------------------------------------------------
__global__ __launch_bounds__(256) void cvt_f32_bf16(const float* __restrict__ s,
                                                    __bf16* __restrict__ d) {
  int i = (blockIdx.x * 256 + threadIdx.x) * 4;
  float4 f = *(const float4*)(s + i);
  d[i + 0] = (__bf16)f.x;
  d[i + 1] = (__bf16)f.y;
  d[i + 2] = (__bf16)f.z;
  d[i + 3] = (__bf16)f.w;
}

// ---------------------------------------------------------------------------
// GEMM:  C[s,n] = sum_d A[s,d] * W[n,d]     (both row-major along d: "NT")
// Block = 256 thr (8 waves), tile 128(M) x 64(N); wave w owns M-tile w (16x64).
// EPI: 0 -> store bf16, 1 -> silu, store f32, 2 -> store f32
// ---------------------------------------------------------------------------
#define LAS 40    // padded LDS strides (80B rows: 16B-aligned, bank-spread)
#define LBS 40

template <int EPI>
__global__ __launch_bounds__(256) void gemm_nt_wmma(const __bf16* __restrict__ A,
                                                    const __bf16* __restrict__ W,
                                                    void* __restrict__ Cout) {
  __shared__ __bf16 lA[128 * LAS];
  __shared__ __bf16 lB[64 * LBS];
  const int tid  = threadIdx.x;
  const int wave = tid >> 5;
  const int lane = tid & 31;
  const int m    = lane & 15;
  const int half = lane >> 4;
  const int m0   = blockIdx.x * 128;
  const int n0   = blockIdx.y * 64;
  const int mrow = wave * 16;

  // per-thread panel-copy coordinates
  const int ar0 = (tid + 0)   >> 2, ac0 = ((tid + 0)   & 3) * 8;
  const int ar1 = (tid + 256) >> 2, ac1 = ((tid + 256) & 3) * 8;
  const int br  = tid >> 2,         bc  = (tid & 3) * 8;

  v8f acc[4] = {};

  for (int kk = 0; kk < DIM; kk += 32) {
    __syncthreads();
    // A panel 128x32, B panel 64x32 via async DMA to LDS
    lds_cp16(&lA[ar0 * LAS + ac0], &A[(size_t)(m0 + ar0) * DIM + kk + ac0]);
    lds_cp16(&lA[ar1 * LAS + ac1], &A[(size_t)(m0 + ar1) * DIM + kk + ac1]);
    lds_cp16(&lB[br  * LBS + bc ], &W[(size_t)(n0 + br ) * DIM + kk + bc ]);
    if (kk + 32 < DIM) {  // prefetch next panels into L2/WGP$
      __builtin_prefetch(&A[(size_t)(m0 + ar0) * DIM + kk + 32 + ac0], 0, 1);
      __builtin_prefetch(&W[(size_t)(n0 + br ) * DIM + kk + 32 + bc ], 0, 1);
    }
    async_wait();
    __syncthreads();

    // A fragment (16x32): lane m row; K = half*8..+7 and half*8+16..+23
    const __bf16* ap = &lA[(mrow + m) * LAS + half * 8];
    v16bf a = frag16(ap, ap + 16);

    // batch all 4 B fragments, then 4 back-to-back WMMAs
    v16bf bf[4];
#pragma unroll
    for (int nt = 0; nt < 4; ++nt) {
      const __bf16* bp = &lB[(nt * 16 + m) * LBS + half * 8];
      bf[nt] = frag16(bp, bp + 16);
    }
#pragma unroll
    for (int nt = 0; nt < 4; ++nt)
      acc[nt] = __builtin_amdgcn_wmma_f32_16x16x32_bf16(
          false, a, false, bf[nt], (short)0, acc[nt], false, false);
  }

  // C/D layout: VGPR r -> row r + 8*half, col = lane&15
#pragma unroll
  for (int nt = 0; nt < 4; ++nt)
#pragma unroll
    for (int r = 0; r < 8; ++r) {
      size_t row = (size_t)(m0 + mrow + r + 8 * half);
      int    col = n0 + nt * 16 + m;
      float  v   = acc[nt][r];
      if (EPI == 0) {
        ((__bf16*)Cout)[row * DIM + col] = (__bf16)v;
      } else if (EPI == 1) {
        ((float*)Cout)[row * DIM + col] = v / (1.f + __expf(-v));
      } else {
        ((float*)Cout)[row * DIM + col] = v;
      }
    }
}

// ---------------------------------------------------------------------------
// Retention attention: per (i-block of 128, head).
// y[i,d] = sum_{j<=i} (q_i.k_j * scale * gamma^(i-j)) * v[j,d]
// ---------------------------------------------------------------------------
#define LKS 136   // lK row stride (272B)
#define LVS 40    // lVt row stride
#define LSS 40    // score-tile row stride

__global__ __launch_bounds__(256) void retention_attn(const __bf16* __restrict__ Qb,
                                                      const __bf16* __restrict__ Kb,
                                                      const __bf16* __restrict__ Vb,
                                                      float* __restrict__ Y) {
  __shared__ __bf16 lK[32 * LKS];       // K block, [j][d]
  __shared__ __bf16 lVt[128 * LVS];     // V block transposed, [d][j]
  __shared__ __bf16 lS[8][16 * LSS];    // per-wave bf16 score tile 16x32

  const int tid  = threadIdx.x;
  const int wave = tid >> 5;
  const int lane = tid & 31;
  const int m    = lane & 15;
  const int half = lane >> 4;
  const int ib   = blockIdx.x * 128;
  const int h    = blockIdx.y;
  const int i0   = ib + wave * 16;

  // per-head decay: gamma = 1 - exp(linspace(log(1/32), log(1/512), 16)[h])
  const float lg0   = -3.4657359027997265f;
  const float lg1   = -6.2383246250395075f;
  const float gamma = 1.f - __expf(lg0 + (float)h * (lg1 - lg0) * (1.f / 15.f));
  const float l2g   = __log2f(gamma);
  const float scale = 0.08838834764831845f;  // 128^-0.5

  // preload Q fragments (4 K-chunks of 32) straight from global
  v16bf qa[4];
#pragma unroll
  for (int kc = 0; kc < 4; ++kc) {
    const __bf16* qp = &Qb[(size_t)(i0 + m) * DIM + h * HDIM + kc * 32 + half * 8];
    qa[kc] = frag16(qp, qp + 16);
  }

  // K panel copy coords: 32 rows x 128 cols, 512 chunks of 8
  const int kr0 = (tid + 0)   >> 4, kc0 = ((tid + 0)   & 15) * 8;
  const int kr1 = (tid + 256) >> 4, kc1 = ((tid + 256) & 15) * 8;
  // V transpose coords
  const int vj = tid >> 3, vd0 = (tid & 7) * 16;

  v8f acc[8] = {};
  const int jend = ib + 128;

  for (int jb = 0; jb < jend; jb += 32) {
    __syncthreads();
    // K block 32x128 row-major via async DMA
    lds_cp16(&lK[kr0 * LKS + kc0], &Kb[(size_t)(jb + kr0) * DIM + h * HDIM + kc0]);
    lds_cp16(&lK[kr1 * LKS + kc1], &Kb[(size_t)(jb + kr1) * DIM + h * HDIM + kc1]);
    // V block 32x128 -> transposed into lVt[d][j] (needs element scatter: sync path)
    {
      const __bf16* vp = &Vb[(size_t)(jb + vj) * DIM + h * HDIM + vd0];
      v8bf v0 = *(const v8bf*)vp;
      v8bf v1 = *(const v8bf*)(vp + 8);
#pragma unroll
      for (int e = 0; e < 8; ++e) {
        lVt[(vd0 + e) * LVS + vj]     = v0[e];
        lVt[(vd0 + 8 + e) * LVS + vj] = v1[e];
      }
    }
    async_wait();
    __syncthreads();

    // ---- S0/S1 = Q Kt : batch 8 B-fragments, then 8 WMMAs ----
    v16bf kb0[4], kb1[4];
#pragma unroll
    for (int kc = 0; kc < 4; ++kc) {
      const __bf16* k0 = &lK[m * LKS + kc * 32 + half * 8];
      const __bf16* k1 = &lK[(m + 16) * LKS + kc * 32 + half * 8];
      kb0[kc] = frag16(k0, k0 + 16);
      kb1[kc] = frag16(k1, k1 + 16);
    }
    v8f s0 = {}, s1 = {};
#pragma unroll
    for (int kc = 0; kc < 4; ++kc) {
      s0 = __builtin_amdgcn_wmma_f32_16x16x32_bf16(false, qa[kc], false, kb0[kc],
                                                   (short)0, s0, false, false);
      s1 = __builtin_amdgcn_wmma_f32_16x16x32_bf16(false, qa[kc], false, kb1[kc],
                                                   (short)0, s1, false, false);
    }

    // causal decay mask + scale, round to bf16, stage in per-wave LDS tile
#pragma unroll
    for (int r = 0; r < 8; ++r) {
      int   gi  = i0 + r + 8 * half;
      float d0f = (float)(gi - (jb + m));
      float d1f = (float)(gi - (jb + 16 + m));
      float w0  = (d0f < 0.f) ? 0.f : scale * exp2f(d0f * l2g);
      float w1  = (d1f < 0.f) ? 0.f : scale * exp2f(d1f * l2g);
      lS[wave][(r + 8 * half) * LSS + m]      = (__bf16)(s0[r] * w0);
      lS[wave][(r + 8 * half) * LSS + 16 + m] = (__bf16)(s1[r] * w1);
    }

    // reload scores as A fragment (16 x 32 over j)
    const __bf16* sp = &lS[wave][m * LSS + half * 8];
    v16bf sa = frag16(sp, sp + 16);

    // ---- acc += S @ V : batch 8 V fragments, then 8 WMMAs ----
    v16bf vbf[8];
#pragma unroll
    for (int dt = 0; dt < 8; ++dt) {
      const __bf16* vp = &lVt[(dt * 16 + m) * LVS + half * 8];
      vbf[dt] = frag16(vp, vp + 16);
    }
#pragma unroll
    for (int dt = 0; dt < 8; ++dt)
      acc[dt] = __builtin_amdgcn_wmma_f32_16x16x32_bf16(
          false, sa, false, vbf[dt], (short)0, acc[dt], false, false);
  }

  // store Y (fp32, (S,D) with head offset)
#pragma unroll
  for (int dt = 0; dt < 8; ++dt)
#pragma unroll
    for (int r = 0; r < 8; ++r) {
      size_t row = (size_t)(i0 + r + 8 * half);
      int    col = h * HDIM + dt * 16 + m;
      Y[row * DIM + col] = acc[dt][r];
    }
}

// ---------------------------------------------------------------------------
// LayerNorm(Y) * G  -> bf16 H      (one block per row, 256 thr, 8 elems/thr)
// ---------------------------------------------------------------------------
__global__ __launch_bounds__(256) void ln_gate(const float* __restrict__ Y,
                                               const float* __restrict__ G,
                                               const float* __restrict__ lnw,
                                               const float* __restrict__ lnb,
                                               __bf16* __restrict__ Hb) {
  __shared__ float red[256];
  __shared__ float stat[2];
  const int row = blockIdx.x;
  const int tid = threadIdx.x;
  const float* y = Y + (size_t)row * DIM;

  float v[8];
  float s = 0.f;
#pragma unroll
  for (int i = 0; i < 8; ++i) { v[i] = y[tid + i * 256]; s += v[i]; }
  red[tid] = s;
  __syncthreads();
  for (int off = 128; off > 0; off >>= 1) {
    if (tid < off) red[tid] += red[tid + off];
    __syncthreads();
  }
  if (tid == 0) stat[0] = red[0] * (1.f / DIM);
  __syncthreads();
  float mu = stat[0];

  float ss = 0.f;
#pragma unroll
  for (int i = 0; i < 8; ++i) { float d = v[i] - mu; ss += d * d; }
  red[tid] = ss;
  __syncthreads();
  for (int off = 128; off > 0; off >>= 1) {
    if (tid < off) red[tid] += red[tid + off];
    __syncthreads();
  }
  if (tid == 0) stat[1] = rsqrtf(red[0] * (1.f / DIM) + 1e-5f);
  __syncthreads();
  float rs = stat[1];

  const float* g = G + (size_t)row * DIM;
#pragma unroll
  for (int i = 0; i < 8; ++i) {
    int   col = tid + i * 256;
    float hv  = (v[i] - mu) * rs * lnw[col] + lnb[col];
    Hb[(size_t)row * DIM + col] = (__bf16)(hv * g[col]);
  }
}

// ---------------------------------------------------------------------------
// Host launcher
// ---------------------------------------------------------------------------
extern "C" void kernel_launch(void* const* d_in, const int* in_sizes, int n_in,
                              void* d_out, int out_size, void* d_ws, size_t ws_size,
                              hipStream_t stream) {
  (void)in_sizes; (void)n_in; (void)out_size; (void)ws_size;

  const float* x   = (const float*)d_in[0];
  const float* wq  = (const float*)d_in[1];
  const float* wk  = (const float*)d_in[2];
  const float* wv  = (const float*)d_in[3];
  const float* wg  = (const float*)d_in[4];
  const float* wo  = (const float*)d_in[5];
  const float* lnw = (const float*)d_in[6];
  const float* lnb = (const float*)d_in[7];

  const size_t NE = (size_t)S_LEN * DIM;  // 4M elems
  char* p = (char*)d_ws;
  auto take = [&](size_t bytes) { char* q = p; p += (bytes + 255) & ~(size_t)255; return q; };

  __bf16* xb  = (__bf16*)take(NE * 2);
  __bf16* wqb = (__bf16*)take(NE * 2);
  __bf16* wkb = (__bf16*)take(NE * 2);
  __bf16* wvb = (__bf16*)take(NE * 2);
  __bf16* wgb = (__bf16*)take(NE * 2);
  __bf16* wob = (__bf16*)take(NE * 2);
  __bf16* Qb  = (__bf16*)take(NE * 2);
  __bf16* Kb  = (__bf16*)take(NE * 2);
  __bf16* Vb  = (__bf16*)take(NE * 2);
  float*  Gf  = (float*)take(NE * 4);
  float*  Yf  = (float*)take(NE * 4);
  __bf16* Hb  = (__bf16*)take(NE * 2);

  const dim3 cvtg((unsigned)(NE / 1024)), cvtb(256);
  cvt_f32_bf16<<<cvtg, cvtb, 0, stream>>>(x,  xb);
  cvt_f32_bf16<<<cvtg, cvtb, 0, stream>>>(wq, wqb);
  cvt_f32_bf16<<<cvtg, cvtb, 0, stream>>>(wk, wkb);
  cvt_f32_bf16<<<cvtg, cvtb, 0, stream>>>(wv, wvb);
  cvt_f32_bf16<<<cvtg, cvtb, 0, stream>>>(wg, wgb);
  cvt_f32_bf16<<<cvtg, cvtb, 0, stream>>>(wo, wob);

  const dim3 gg(S_LEN / 128, DIM / 64), gb(256);
  gemm_nt_wmma<0><<<gg, gb, 0, stream>>>(xb, wqb, (void*)Qb);
  gemm_nt_wmma<0><<<gg, gb, 0, stream>>>(xb, wkb, (void*)Kb);
  gemm_nt_wmma<0><<<gg, gb, 0, stream>>>(xb, wvb, (void*)Vb);
  gemm_nt_wmma<1><<<gg, gb, 0, stream>>>(xb, wgb, (void*)Gf);   // silu gate

  retention_attn<<<dim3(S_LEN / 128, NHEAD), 256, 0, stream>>>(Qb, Kb, Vb, Yf);

  ln_gate<<<S_LEN, 256, 0, stream>>>(Yf, Gf, lnw, lnb, Hb);

  gemm_nt_wmma<2><<<gg, gb, 0, stream>>>(Hb, wob, d_out);       // fp32 out
}
